// dMasifWrapper_67018669687528
// MI455X (gfx1250) — matrix operand order repaired and checked by reading, not compiled, not run on hardware
//
#include <hip/hip_runtime.h>

// ---------------------------------------------------------------------------
// dMaSIF forward for MI455X (gfx1250, wave32).
// ~11 GFLOP, ~10 MB resident -> lives in L2; compute-bound.
// Pairwise S/T accumulations and all dense MLPs run on V_WMMA_F32_16X16X32_F16
// (f16 in / f32 accumulate). Pairwise exp-window generation stays on VALU.
// f-tiles are moved to LDS with GLOBAL_LOAD_ASYNC_TO_LDS_B128 (ASYNCcnt),
// overlapped with A-fragment generation.
// ---------------------------------------------------------------------------

#define NPTS   2048
#define DIN    128
#define ICH    138          // DIM_IN + 2*len(SCALES)
#define ICP    160          // ICH padded to multiple of 32 for WMMA K
#define HCH    128
#define CUTS   8
#define NGROUP 4
#define GN_EPS 1e-5f
#define RADIUS 9.0f

#ifndef HAVE_ASYNC_LDS
#if defined(__has_builtin)
#if __has_builtin(__builtin_amdgcn_global_load_async_to_lds_b128)
#define HAVE_ASYNC_LDS 1
#endif
#endif
#endif
#ifndef HAVE_ASYNC_LDS
#define HAVE_ASYNC_LDS 0
#endif

#if defined(__has_builtin)
#if __has_builtin(__builtin_amdgcn_s_wait_asynccnt)
#define WAIT_ASYNC0() __builtin_amdgcn_s_wait_asynccnt(0)
#endif
#endif
#ifndef WAIT_ASYNC0
#define WAIT_ASYNC0() asm volatile("s_wait_asynccnt 0x0" ::: "memory")
#endif

typedef __attribute__((ext_vector_type(16))) _Float16 v16h;
typedef __attribute__((ext_vector_type(8)))  float    v8f;
typedef int v4i __attribute__((vector_size(16)));      // matches builtin param
typedef v4i __attribute__((address_space(1)))* gv4i_p; // global (AS1) int4*
typedef v4i __attribute__((address_space(3)))* lv4i_p; // LDS (AS3) int4*

__device__ __forceinline__ float leakyf(float x) { return x >= 0.f ? x : 0.2f * x; }

__device__ __forceinline__ void tangent_frame(float nx, float ny, float nz,
                                              float* u, float* v) {
  float s = (nz >= 0.f) ? 1.f : -1.f;
  float a = -1.f / (s + nz);
  float b = nx * ny * a;
  u[0] = 1.f + s * nx * nx * a; u[1] = s * b;          u[2] = -s * nx;
  v[0] = b;                     v[1] = s + ny * ny * a; v[2] = -ny;
}

// WMMA fragment-layout helpers (CDNA5 ISA 7.12.2, 16-bit A 16x32 / B 32x16).
__device__ __forceinline__ int a_frag_off(int m, int k) {
  int kk = k & 15;
  int lane = m + ((kk >= 8) ? 16 : 0);
  int slot = ((k >> 4) << 3) + (kk & 7);
  return lane * 16 + slot;
}
__device__ __forceinline__ int b_frag_off(int k, int n) {
  int lane = (n & 15) + (((k & 15) >= 8) ? 16 : 0);
  return lane * 16 + ((k >> 4) << 3) + (k & 7);
}

// ------------------------- curvature stage (VALU) ---------------------------

__global__ void k_smooth_normals(const float* __restrict__ verts,
                                 const float* __restrict__ vn,
                                 const int* __restrict__ batch,
                                 float inv2s2, float* __restrict__ ns) {
  int i = blockIdx.x * blockDim.x + threadIdx.x;
  if (i >= NPTS) return;
  float px = verts[i*3], py = verts[i*3+1], pz = verts[i*3+2];
  int bi = batch[i];
  float ux = 0.f, uy = 0.f, uz = 0.f;
  for (int j = 0; j < NPTS; ++j) {
    float dx = verts[j*3] - px, dy = verts[j*3+1] - py, dz = verts[j*3+2] - pz;
    float d2 = dx*dx + dy*dy + dz*dz;
    float k = (batch[j] == bi) ? __expf(-d2 * inv2s2) : 0.f;
    ux += k * vn[j*3]; uy += k * vn[j*3+1]; uz += k * vn[j*3+2];
  }
  float nrm = fmaxf(sqrtf(ux*ux + uy*uy + uz*uz), 1e-12f);
  ns[i*3] = ux / nrm; ns[i*3+1] = uy / nrm; ns[i*3+2] = uz / nrm;
}

__global__ void k_curv(const float* __restrict__ verts,
                       const int* __restrict__ batch,
                       const float* __restrict__ ns,
                       float inv2s2, int sidx, float* __restrict__ curv) {
  int i = blockIdx.x * blockDim.x + threadIdx.x;
  if (i >= NPTS) return;
  float px = verts[i*3], py = verts[i*3+1], pz = verts[i*3+2];
  float nix = ns[i*3], niy = ns[i*3+1], niz = ns[i*3+2];
  float u[3], v[3];
  tangent_frame(nix, niy, niz, u, v);
  int bi = batch[i];
  float pp00=0,pp01=0,pp10=0,pp11=0, pq00=0,pq01=0,pq10=0,pq11=0;
  for (int j = 0; j < NPTS; ++j) {
    float dx = verts[j*3] - px, dy = verts[j*3+1] - py, dz = verts[j*3+2] - pz;
    float d2 = dx*dx + dy*dy + dz*dz;
    float njx = ns[j*3], njy = ns[j*3+1], njz = ns[j*3+2];
    float nd = nix*njx + niy*njy + niz*njz;
    float t = 2.f - nd;
    float w = (batch[j] == bi) ? __expf(-d2 * t * t * inv2s2) : 0.f;
    float P0 = u[0]*dx + u[1]*dy + u[2]*dz;
    float P1 = v[0]*dx + v[1]*dy + v[2]*dz;
    float ex = njx - nix, ey = njy - niy, ez = njz - niz;
    float Q0 = u[0]*ex + u[1]*ey + u[2]*ez;
    float Q1 = v[0]*ex + v[1]*ey + v[2]*ez;
    pp00 += w*P0*P0; pp01 += w*P0*P1; pp10 += w*P1*P0; pp11 += w*P1*P1;
    pq00 += w*P0*Q0; pq01 += w*P0*Q1; pq10 += w*P1*Q0; pq11 += w*P1*Q1;
  }
  pp00 += 0.01f; pp11 += 0.01f;
  float id = 1.f / (pp00*pp11 - pp01*pp10);
  float S00 = ( pp11*pq00 - pp01*pq10) * id;
  float S01 = ( pp11*pq01 - pp01*pq11) * id;
  float S10 = (-pp10*pq00 + pp00*pq10) * id;
  float S11 = (-pp10*pq01 + pp00*pq11) * id;
  float tr = S00 + S11, dd = S00*S11 - S01*S10;
  curv[i*10 + 2*sidx]     = fminf(1.f, fmaxf(-1.f, tr));
  curv[i*10 + 2*sidx + 1] = fminf(1.f, fmaxf(-1.f, dd));
}

__global__ void k_feats(const float* __restrict__ x, const float* __restrict__ curv,
                        float* __restrict__ featsP) {
  int idx = blockIdx.x * blockDim.x + threadIdx.x;
  if (idx >= NPTS * ICP) return;
  int n = idx / ICP, c = idx % ICP;
  float val = 0.f;
  if (c < DIN) val = x[n*DIN + c];
  else if (c < ICH) val = curv[n*10 + (c - DIN)];
  featsP[idx] = val;   // zero-padded K tail for WMMA
}

// ------------------------- generic WMMA GEMM --------------------------------
// C[N x 128] = act(A[N x Kp] @ W[128 x K]^T + bias) (+ resid). Kp % 32 == 0,
// A zero-padded past K. One block = 16 rows x 128 cols; 8 waves, 1 h-tile each.

__global__ void __launch_bounds__(256)
k_gemm(const float* __restrict__ A, int K, int Kp,
       const float* __restrict__ W, const float* __restrict__ bias,
       float* __restrict__ C, int act, const float* __restrict__ resid) {
  __shared__ alignas(32) _Float16 Ah[512];
  __shared__ alignas(32) _Float16 Bh[8][512];
  int tid = threadIdx.x, lane = tid & 31, wv = tid >> 5;
  int it = blockIdx.x;
  v8f acc = {};
  int chunks = Kp >> 5;
  int am = (tid * 2) >> 5, ak = (tid * 2) & 31;        // A staging: float2/thread
  for (int kc = 0; kc < chunks; ++kc) {
    const float2 av = *reinterpret_cast<const float2*>(
        &A[(it*16 + am) * Kp + kc*32 + ak]);
    Ah[a_frag_off(am, ak)]     = (_Float16)av.x;
    Ah[a_frag_off(am, ak + 1)] = (_Float16)av.y;
    for (int e = tid; e < 4096; e += 256) {            // W chunk 128x32
      int k = e & 31, hc = e >> 5;
      int kg = kc*32 + k;
      float val = (kg < K) ? W[hc*K + kg] : 0.f;
      Bh[hc >> 4][b_frag_off(k, hc & 15)] = (_Float16)val;
    }
    __syncthreads();
    v16h a = *reinterpret_cast<const v16h*>(&Ah[lane * 16]);
    v16h b = *reinterpret_cast<const v16h*>(&Bh[wv][lane * 16]);
    acc = __builtin_amdgcn_wmma_f32_16x16x32_f16(false, a, false, b,
                                                 (short)0, acc, false, false);
    __syncthreads();
  }
  for (int r = 0; r < 8; ++r) {
    int m = r + ((lane >= 16) ? 8 : 0);
    int n = (wv << 4) + (lane & 15);
    int row = it*16 + m;
    float val = acc[r] + bias[n];
    if (act == 1)      val = leakyf(val);
    else if (act == 2) val = fmaxf(val, 0.f);
    if (resid) val += resid[row * HCH + n];
    C[row * HCH + n] = val;
  }
}

// ------------------------- small VALU kernels -------------------------------

__global__ void k_weights(const float* __restrict__ h, const float* __restrict__ w2,
                          const float* __restrict__ b2, float* __restrict__ wts) {
  int i = blockIdx.x * blockDim.x + threadIdx.x;
  if (i >= NPTS) return;
  float acc = b2[0];
  for (int c = 0; c < HCH; ++c) acc += h[i*HCH + c] * w2[c];
  wts[i] = acc;
}

__global__ void k_nuv(const float* __restrict__ verts, const float* __restrict__ vn,
                      const int* __restrict__ batch, const float* __restrict__ wts,
                      float* __restrict__ nuv) {
  int i = blockIdx.x * blockDim.x + threadIdx.x;
  if (i >= NPTS) return;
  const float invR = 1.f / RADIUS;
  float px = verts[i*3]*invR, py = verts[i*3+1]*invR, pz = verts[i*3+2]*invR;
  float nx = vn[i*3], ny = vn[i*3+1], nz = vn[i*3+2];
  float u[3], v[3];
  tangent_frame(nx, ny, nz, u, v);
  int bi = batch[i];
  float ov0 = 0.f, ov1 = 0.f;
  for (int j = 0; j < NPTS; ++j) {
    float dx = verts[j*3]*invR - px, dy = verts[j*3+1]*invR - py, dz = verts[j*3+2]*invR - pz;
    float d2 = dx*dx + dy*dy + dz*dz;
    float nd = nx*vn[j*3] + ny*vn[j*3+1] + nz*vn[j*3+2];
    float t = 2.f - nd;
    float win = (batch[j] == bi) ? __expf(-d2 * t * t) : 0.f;
    float ww = win * wts[j];
    ov0 += ww * (u[0]*dx + u[1]*dy + u[2]*dz);
    ov1 += ww * (v[0]*dx + v[1]*dy + v[2]*dz);
  }
  ov0 += 1e-5f; ov1 += 1e-5f;
  float nrm = fmaxf(sqrtf(ov0*ov0 + ov1*ov1), 1e-12f);
  float ex = ov0 / nrm, ey = ov1 / nrm;
  nuv[i*9+0] = nx; nuv[i*9+1] = ny; nuv[i*9+2] = nz;
  nuv[i*9+3] =  ex*u[0] + ey*v[0]; nuv[i*9+4] =  ex*u[1] + ey*v[1]; nuv[i*9+5] =  ex*u[2] + ey*v[2];
  nuv[i*9+6] = -ey*u[0] + ex*v[0]; nuv[i*9+7] = -ey*u[1] + ex*v[1]; nuv[i*9+8] = -ey*u[2] + ex*v[2];
}

__global__ void k_gn_reduce(const float* __restrict__ f, float* __restrict__ stats) {
  __shared__ float s1[256], s2[256];
  int g = blockIdx.x;
  float sum = 0.f, sq = 0.f;
  for (int idx = threadIdx.x; idx < 32 * NPTS; idx += 256) {
    float val = f[(idx >> 5) * HCH + g*32 + (idx & 31)];
    sum += val; sq += val * val;
  }
  s1[threadIdx.x] = sum; s2[threadIdx.x] = sq;
  __syncthreads();
  for (int s = 128; s > 0; s >>= 1) {
    if (threadIdx.x < s) { s1[threadIdx.x] += s1[threadIdx.x + s];
                           s2[threadIdx.x] += s2[threadIdx.x + s]; }
    __syncthreads();
  }
  if (threadIdx.x == 0) {
    const float inv = 1.f / (32.f * NPTS);
    float mu = s1[0] * inv;
    float var = s2[0] * inv - mu * mu;
    stats[g*2]     = mu;
    stats[g*2 + 1] = rsqrtf(var + GN_EPS);
  }
}

// group-norm apply; optionally also emits an f16 copy (feed for WMMA conv).
__global__ void k_gn_apply(const float* __restrict__ f, const float* __restrict__ stats,
                           const float* __restrict__ gamma, const float* __restrict__ beta,
                           float* __restrict__ out, _Float16* __restrict__ out16) {
  int idx = blockIdx.x * blockDim.x + threadIdx.x;
  if (idx >= NPTS * HCH) return;
  int c = idx % HCH, g = c >> 5;
  float val = (f[idx] - stats[g*2]) * stats[g*2 + 1] * gamma[c] + beta[c];
  out[idx] = val;
  if (out16) out16[idx] = (_Float16)val;
}

// ------------------- flagship pairwise conv (WMMA) --------------------------
// S[i,c,h] = sum_j (w_ij*R_ijc) f_jh ; T[i,h] = sum_j w_ij f_jh
// f16 f-tiles are DMA'd to LDS asynchronously (ASYNCcnt) while the block
// generates the 9 A-fragments (8 cuts + plain w) on VALU; 8 waves each hold one
// h-tile and run 9 v_wmma per j-tile. Epilogue fuses A2.S + b2.T.

__global__ void __launch_bounds__(256)
k_conv(const float* __restrict__ verts, const int* __restrict__ batch,
       const float* __restrict__ nuv, const _Float16* __restrict__ f16,
       const float* __restrict__ A1, const float* __restrict__ b1,
       const float* __restrict__ A2, const float* __restrict__ b2,
       float* __restrict__ out) {
  __shared__ alignas(16) _Float16 rawF[32 * 128];       // 8 KB raw f16 tile
  __shared__ alignas(32) _Float16 Ah[CUTS + 1][512];
  __shared__ alignas(32) _Float16 Bh[8][512];
  __shared__ float sA1[24], sb1[8];
  int tid = threadIdx.x, lane = tid & 31, wv = tid >> 5;
  int it = blockIdx.x;
  if (tid < 24) sA1[tid] = A1[tid];
  if (tid < 8)  sb1[tid] = b1[tid];
  const float scale = 1.f / (1.41421356237f * RADIUS);

  // i-side data is invariant over the whole j loop for this thread's two pairs
  int m0 = (tid * 2) >> 5;
  int i0 = it * 16 + m0;
  float pix = verts[i0*3] * scale, piy = verts[i0*3+1] * scale, piz = verts[i0*3+2] * scale;
  float n0x = nuv[i0*9+0], n0y = nuv[i0*9+1], n0z = nuv[i0*9+2];
  float t0x = nuv[i0*9+3], t0y = nuv[i0*9+4], t0z = nuv[i0*9+5];
  float b0x = nuv[i0*9+6], b0y = nuv[i0*9+7], b0z = nuv[i0*9+8];
  int bi0 = batch[i0];

  v8f acc[CUTS + 1] = {};
  __syncthreads();

  for (int jt = 0; jt < NPTS / 32; ++jt) {
    // --- kick off async f16 tile copy to LDS (overlaps A generation) -------
#if defined(__AMDGCN__) && HAVE_ASYNC_LDS
    for (int e = tid; e < 512; e += 256) {
      gv4i_p g = (gv4i_p)(unsigned long long)(
          reinterpret_cast<unsigned long long>(f16 + (size_t)jt * 4096) + (size_t)e * 16);
      lv4i_p l = (lv4i_p)(unsigned)(
          (unsigned)(unsigned long long)reinterpret_cast<unsigned long long>(rawF) + e * 16);
      __builtin_amdgcn_global_load_async_to_lds_b128(g, l, 0, 0);
    }
#else
    for (int e = tid; e < 512; e += 256)
      reinterpret_cast<int4*>(rawF)[e] =
          reinterpret_cast<const int4*>(f16 + (size_t)jt * 4096)[e];
#endif

    // --- generate 9 A-fragment tiles for this (16 x 32) pair tile ----------
    for (int p = 0; p < 2; ++p) {
      int kj = ((tid * 2) & 31) + p;
      int j = jt * 32 + kj;
      float dx = verts[j*3]   * scale - pix;
      float dy = verts[j*3+1] * scale - piy;
      float dz = verts[j*3+2] * scale - piz;
      float d2 = dx*dx + dy*dy + dz*dz;
      float nd = n0x*nuv[j*9] + n0y*nuv[j*9+1] + n0z*nuv[j*9+2];
      float t = 2.f - nd;
      float w = (bi0 == batch[j]) ? __expf(-d2 * t * t) : 0.f;
      float X0 = n0x*dx + n0y*dy + n0z*dz;
      float X1 = t0x*dx + t0y*dy + t0z*dz;
      float X2 = b0x*dx + b0y*dy + b0z*dz;
      int off = a_frag_off(m0, kj);
      for (int c = 0; c < CUTS; ++c) {
        float r = fmaxf(sA1[c*3]*X0 + sA1[c*3+1]*X1 + sA1[c*3+2]*X2 + sb1[c], 0.f);
        Ah[c][off] = (_Float16)(w * r);
      }
      Ah[CUTS][off] = (_Float16)w;
    }

#if defined(__AMDGCN__) && HAVE_ASYNC_LDS
    WAIT_ASYNC0();
#endif
    __syncthreads();                    // rawF + Ah visible to all

    // --- swizzle raw tile into B fragments ---------------------------------
    for (int e = tid; e < 4096; e += 256) {
      int k = e & 31, hc = e >> 5;
      Bh[hc >> 4][b_frag_off(k, hc & 15)] = rawF[k * 128 + hc];
    }
    __syncthreads();                    // Bh ready

    v16h b = *reinterpret_cast<const v16h*>(&Bh[wv][lane * 16]);
    for (int c = 0; c <= CUTS; ++c) {
      v16h a = *reinterpret_cast<const v16h*>(&Ah[c][lane * 16]);
      acc[c] = __builtin_amdgcn_wmma_f32_16x16x32_f16(false, a, false, b,
                                                      (short)0, acc[c], false, false);
    }
    __syncthreads();                    // done reading before next overwrite
  }

  for (int r = 0; r < 8; ++r) {
    int m = r + ((lane >= 16) ? 8 : 0);
    int h = (wv << 4) + (lane & 15);
    float o = b2[h] * acc[CUTS][r];
    for (int c = 0; c < CUTS; ++c) o += A2[h*CUTS + c] * acc[c][r];
    out[(it*16 + m) * HCH + h] = o;
  }
}

// ----------------------------- launcher -------------------------------------

extern "C" void kernel_launch(void* const* d_in, const int* in_sizes, int n_in,
                              void* d_out, int out_size, void* d_ws, size_t ws_size,
                              hipStream_t stream) {
  // setup_inputs order, params flattened in jax pytree-leaf (sorted-key) order.
  const float* verts    = (const float*)d_in[0];
  const float* vnormals = (const float*)d_in[1];
  const float* x        = (const float*)d_in[2];
  const int*   batch    = (const int*)  d_in[3];
  const float* A1       = (const float*)d_in[4];
  const float* A2       = (const float*)d_in[5];
  const float* b1       = (const float*)d_in[6];
  const float* b2       = (const float*)d_in[7];
  const float* gn_in_b  = (const float*)d_in[8];
  const float* gn_in_g  = (const float*)d_in[9];
  const float* gn_out_b = (const float*)d_in[10];
  const float* gn_out_g = (const float*)d_in[11];
  const float* in_b1    = (const float*)d_in[12];
  const float* in_b2    = (const float*)d_in[13];
  const float* in_w1    = (const float*)d_in[14];
  const float* in_w2    = (const float*)d_in[15];
  const float* ll_b1    = (const float*)d_in[16];
  const float* ll_b2    = (const float*)d_in[17];
  const float* ll_w1    = (const float*)d_in[18];
  const float* ll_w2    = (const float*)d_in[19];
  const float* lt_b     = (const float*)d_in[20];
  const float* lt_w     = (const float*)d_in[21];
  const float* out_b1   = (const float*)d_in[22];
  const float* out_b2   = (const float*)d_in[23];
  const float* out_w1   = (const float*)d_in[24];
  const float* out_w2   = (const float*)d_in[25];
  const float* os_b1    = (const float*)d_in[26];
  const float* os_b2    = (const float*)d_in[27];
  const float* os_w1    = (const float*)d_in[28];
  const float* os_w2    = (const float*)d_in[29];
  (void)in_sizes; (void)n_in; (void)out_size; (void)ws_size;

  float* ws     = (float*)d_ws;
  float* curv   = ws;                    // N*10
  float* featsP = curv   + NPTS * 10;    // N*160 (zero-padded)
  float* nsbuf  = featsP + NPTS * ICP;   // N*3
  float* wts    = nsbuf  + NPTS * 3;     // N
  float* nuv    = wts    + NPTS;         // N*9
  float* bufA   = nuv    + NPTS * 9;     // N*128
  float* bufB   = bufA   + NPTS * HCH;
  float* bufC   = bufB   + NPTS * HCH;
  float* bufD   = bufC   + NPTS * HCH;
  float* stats  = bufD   + NPTS * HCH;   // 8
  _Float16* h16 = (_Float16*)(stats + 8);// N*128 f16 (16B-aligned offset)

  dim3 blk(256);
  dim3 gridN((NPTS + 255) / 256);
  dim3 gridT(NPTS / 16);

  // curvature features (5 scales)
  const float scales[5] = {1.f, 2.f, 3.f, 5.f, 10.f};
  for (int s = 0; s < 5; ++s) {
    float inv2s2 = 1.f / (2.f * scales[s] * scales[s]);
    k_smooth_normals<<<gridN, blk, 0, stream>>>(verts, vnormals, batch, inv2s2, nsbuf);
    k_curv<<<gridN, blk, 0, stream>>>(verts, batch, nsbuf, inv2s2, s, curv);
  }
  k_feats<<<(NPTS * ICP + 255) / 256, blk, 0, stream>>>(x, curv, featsP);

  // orientation scores -> weights -> nuv frames
  k_gemm<<<gridT, blk, 0, stream>>>(featsP, ICH, ICP, os_w1, os_b1, bufA, 1, nullptr);
  k_weights<<<gridN, blk, 0, stream>>>(bufA, os_w2, os_b2, wts);
  k_nuv<<<gridN, blk, 0, stream>>>(verts, vnormals, batch, wts, nuv);

  // input MLP + group norm -> f (f32 + f16 copy for the conv)
  k_gemm<<<gridT, blk, 0, stream>>>(featsP, ICH, ICP, in_w1, in_b1, bufB, 1, nullptr);
  k_gemm<<<gridT, blk, 0, stream>>>(bufB, HCH, HCH, in_w2, in_b2, bufC, 1, nullptr);
  k_gn_reduce<<<NGROUP, blk, 0, stream>>>(bufC, stats);
  k_gn_apply<<<(NPTS * HCH + 255) / 256, blk, 0, stream>>>(bufC, stats, gn_in_g, gn_in_b, bufA, h16);

  // quasi-geodesic conv (WMMA + async-LDS tiles)
  k_conv<<<gridT, blk, 0, stream>>>(verts, batch, nuv, h16, A1, b1, A2, b2, bufB);

  // output MLP + group norm
  k_gemm<<<gridT, blk, 0, stream>>>(bufB, HCH, HCH, out_w1, out_b1, bufC, 1, nullptr);
  k_gemm<<<gridT, blk, 0, stream>>>(bufC, HCH, HCH, out_w2, out_b2, bufD, 1, nullptr);
  k_gn_reduce<<<NGROUP, blk, 0, stream>>>(bufD, stats);
  k_gn_apply<<<(NPTS * HCH + 255) / 256, blk, 0, stream>>>(bufD, stats, gn_out_g, gn_out_b, bufB, nullptr);

  // post-linear (ll) and transform residual (lt) -> d_out
  k_gemm<<<gridT, blk, 0, stream>>>(bufB, HCH, HCH, ll_w1, ll_b1, bufC, 2, nullptr);
  k_gemm<<<gridT, blk, 0, stream>>>(bufC, HCH, HCH, ll_w2, ll_b2, bufD, 0, nullptr);
  k_gemm<<<gridT, blk, 0, stream>>>(featsP, ICH, ICP, lt_w, lt_b, (float*)d_out, 0, bufD);
}
